// GETLayer_12635793785237
// MI455X (gfx1250) — matrix operand order repaired and checked by reading, not compiled, not run on hardware
//
#include <hip/hip_runtime.h>
#include <math.h>

// ---- problem constants (from the reference) ----
#define N_NODES 20000
#define EDGES   160000
#define DIM     256
#define HEADS   4
#define ZDIM    64
#define KMEM    32
#define NC1     704      // GEMM1 output cols: Q2(256) | K2(256) | mem-scores(128) | motif-scores(16) | pad(48)
#define KPACK   640      // GEMM2 K: l2*msg_dst(256) | l2*msg_src(256) | lm*Am(128)
#define BETA_MAXV 5.0f
#define DAMP      0.9999f
#define EPS_LN    1e-5f

typedef __attribute__((ext_vector_type(16))) __bf16        v16bf;
typedef __attribute__((ext_vector_type(8)))  float         v8f;
typedef __attribute__((ext_vector_type(4)))  unsigned int  v4u;
typedef __attribute__((ext_vector_type(4)))  float         v4f;

union Frag { v4u u[2]; v16bf v; };

static __device__ __forceinline__ unsigned short f2bf(float f) {
  union { float f; unsigned int u; } v; v.f = f;
  unsigned int r = v.u + 0x7fffu + ((v.u >> 16) & 1u);   // RNE
  return (unsigned short)(r >> 16);
}
static __device__ __forceinline__ float softplus_f(float x) { return log1pf(__expf(x)); }
static __device__ __forceinline__ float gelu_tanh(float x) {
  float x3 = x * x * x;
  return 0.5f * x * (1.f + tanhf(0.7978845608028654f * (x + 0.044715f * x3)));
}
static __device__ __forceinline__ void atomicMaxF(float* a, float v) {
  if (v >= 0.f) atomicMax((int*)a, __float_as_int(v));
  else          atomicMin((unsigned int*)a, __float_as_uint(v));
}
// Async DMA of 16B global -> LDS (CDNA5, tracked by ASYNCcnt). Inline asm per
// CDNA5_HIP.md recommendation (portable across toolchains). ldsaddr = LDS byte
// address (low 32 bits of a generic LDS pointer).
static __device__ __forceinline__ void async_g2l_b128(unsigned int ldsaddr, const void* gaddr) {
  asm volatile("global_load_async_to_lds_b128 %0, %1, off"
               :: "v"(ldsaddr), "v"(gaddr) : "memory");
}

// ------------------------------------------------------------------
// small fills / conversions
// ------------------------------------------------------------------
__global__ void k_fillf(float* __restrict__ p, float v, int n) {
  int i = blockIdx.x * blockDim.x + threadIdx.x;
  if (i < n) p[i] = v;
}

// W_Q2/W_K2 [256 rows x 256] -> bf16 into Wcat1 (row-major over K=d) and
// Wcat2T (transposed: row = output dim d, K = packed index p).
__global__ void k_cvtW(const float* __restrict__ src, unsigned short* __restrict__ Wcat1,
                       unsigned short* __restrict__ Wcat2T, int rowbase) {
  int idx = blockIdx.x * 256 + threadIdx.x;      // 65536
  int p = idx >> 8, d = idx & 255;
  unsigned short b = f2bf(src[idx]);
  int row = rowbase + p;
  Wcat1[(size_t)row * DIM + d]   = b;
  Wcat2T[(size_t)d * KPACK + row] = b;
}

// Km[h,k,z] = sum_d B_mem[k,d] * W_Km[h,z,d]
__global__ void k_km(const float* __restrict__ Bm, const float* __restrict__ Wkm,
                     float* __restrict__ Km) {
  int idx = blockIdx.x * 256 + threadIdx.x;      // 8192
  int h = idx >> 11, rem = idx & 2047, k = rem >> 6, z = rem & 63;
  const float* b = Bm + (size_t)k * DIM;
  const float* w = Wkm + (size_t)(h * ZDIM + z) * DIM;
  float s = 0.f;
  for (int d = 0; d < DIM; d += 4) {
    v4f a = *(const v4f*)(b + d); v4f c = *(const v4f*)(w + d);
    s += a.x * c.x + a.y * c.y + a.z * c.z + a.w * c.w;
  }
  Km[idx] = s;
}

// U_{Q,K}[t,h,d] = sum_p T_tau[t,h,p] * W_{Q3,K3}[h,p,d]; also bf16 rows 640..655 of Wcat1
__global__ void k_u(const float* __restrict__ Ttau, const float* __restrict__ Wq3,
                    const float* __restrict__ Wk3, float* __restrict__ Uq,
                    float* __restrict__ Uk, unsigned short* __restrict__ Wcat1) {
  int idx = blockIdx.x * 256 + threadIdx.x;      // 4096
  int qk = idx >> 11, rem = idx & 2047;
  int t = rem >> 10, hd = rem & 1023, h = hd >> 8, d = hd & 255;
  const float* tt = Ttau + (size_t)(t * HEADS + h) * 256;
  const float* w  = (qk ? Wk3 : Wq3) + (size_t)h * 256 * DIM + d;
  float s = 0.f;
  for (int p = 0; p < 256; ++p) s += tt[p] * w[(size_t)p * DIM];
  (qk ? Uk : Uq)[(t * HEADS + h) * DIM + d] = s;
  Wcat1[(size_t)(640 + qk * 8 + h * 2 + t) * DIM + d] = f2bf(s);
}

// P[h,k,d] = sum_z Km[h,k,z]*W_Qm[h,z,d]  (serves both the score GEMM and grad GEMM)
__global__ void k_m2p(const float* __restrict__ Km, const float* __restrict__ Wqm,
                      unsigned short* __restrict__ Wcat1, unsigned short* __restrict__ Wcat2T) {
  int idx = blockIdx.x * 256 + threadIdx.x;      // 32768
  int hk = idx >> 8, d = idx & 255;
  int h = hk >> 5, k = hk & 31;
  const float* km = Km + (size_t)(h * KMEM + k) * ZDIM;
  const float* w  = Wqm + (size_t)h * ZDIM * DIM + d;
  float s = 0.f;
  for (int z = 0; z < ZDIM; ++z) s += km[z] * w[(size_t)z * DIM];
  unsigned short b = f2bf(s);
  Wcat1[(size_t)(512 + hk) * DIM + d]    = b;
  Wcat2T[(size_t)d * KPACK + 512 + hk]   = b;
}

// ------------------------------------------------------------------
// layernorm -> bf16 activations
// ------------------------------------------------------------------
__global__ __launch_bounds__(256) void k_layernorm(const float* __restrict__ G,
                                                   const float* __restrict__ lg,
                                                   const float* __restrict__ lb,
                                                   unsigned short* __restrict__ gbf) {
  __shared__ float red[256];
  int n = blockIdx.x, d = threadIdx.x;
  float x = G[(size_t)n * DIM + d];
  red[d] = x; __syncthreads();
  for (int s = 128; s; s >>= 1) { if (d < s) red[d] += red[d + s]; __syncthreads(); }
  float mu = red[0] * (1.0f / DIM); __syncthreads();
  float c = x - mu;
  red[d] = c * c; __syncthreads();
  for (int s = 128; s; s >>= 1) { if (d < s) red[d] += red[d + s]; __syncthreads(); }
  float var = red[0] * (1.0f / DIM);
  float y = c * rsqrtf(var + EPS_LN) * lg[d] + lb[d];
  gbf[(size_t)n * DIM + d] = f2bf(y);
}

// ------------------------------------------------------------------
// bf16 WMMA GEMM: C[m,n] = sum_k A[m,k]*B[n,k]; A [M,K], B [NC,K] row-major bf16.
// Block = 8 waves, tile 256Mx64N; each wave computes a 32x64 strip
// (2 A-frags x 4 B-frags = 8 WMMAs per K-tile).
// The 64x32 B tile (shared by all 8 waves) is staged in LDS via a 2-deep
// async-DMA pipeline (global_load_async_to_lds_b128 / s_wait_asynccnt).
// Steady-state loop is branch-free: unconditional s_wait_asynccnt 1 and
// unconditional refill; the last two tiles are peeled as the epilogue.
// All 8 ds_load_b128 for the 4 B-frags are issued before the WMMA group so
// the scheduler can use partial DScnt waits. Rows padded to 80B for banks.
// ------------------------------------------------------------------
#define BSTRIDE 40   // halves per LDS row (80 bytes)

__global__ __launch_bounds__(256) void k_gemm_bf16(
    const unsigned short* __restrict__ A, const unsigned short* __restrict__ B,
    float* __restrict__ C, int M, int K, int NC)
{
  __shared__ unsigned short bs[2][64 * BSTRIDE];
  const int tid  = threadIdx.x;
  const int lane = tid & 31;
  const int wv   = tid >> 5;
  const int l16  = lane & 15;
  const int hi   = lane >> 4;
  const int mbase = blockIdx.x * 256 + wv * 32;
  const int nbase = blockIdx.y * 64;

  // staging ownership: thread -> (row 0..63, 16B chunk 0..3) of the 64x32 tile
  const int srow = tid >> 2;
  const int schk = tid & 3;
  const unsigned short* gB = B + (size_t)(nbase + srow) * K + schk * 8;
  const unsigned int lds0 =
      (unsigned int)(unsigned long long)&bs[0][srow * BSTRIDE + schk * 8];
  const unsigned int lds1 =
      (unsigned int)(unsigned long long)&bs[1][srow * BSTRIDE + schk * 8];

  int ar0 = mbase + l16;        if (ar0 >= M) ar0 = M - 1;  // clamp; never stored
  int ar1 = mbase + 16 + l16;   if (ar1 >= M) ar1 = M - 1;
  const unsigned short* ap0 = A + (size_t)ar0 * K + hi * 8;
  const unsigned short* ap1 = A + (size_t)ar1 * K + hi * 8;

  v8f acc[8];
  for (int j = 0; j < 8; ++j)
    for (int r = 0; r < 8; ++r) acc[j][r] = 0.f;

  // one K-tile of compute: frag loads (batched) + 8 WMMAs from LDS buf (t&1)
  auto compute_tile = [&](int t) {
    const int kt = t << 5;
    Frag af0, af1;
    af0.u[0] = *(const v4u*)(ap0 + kt);        // K = kt + hi*8 + [0..7]
    af0.u[1] = *(const v4u*)(ap0 + kt + 16);   // K = kt + 16 + hi*8 + [0..7]
    af1.u[0] = *(const v4u*)(ap1 + kt);
    af1.u[1] = *(const v4u*)(ap1 + kt + 16);
    if (kt + 32 < K) {
      __builtin_prefetch(ap0 + kt + 32, 0, 0);
      __builtin_prefetch(ap1 + kt + 32, 0, 0);
    }
    const unsigned short* lb = &bs[t & 1][hi * 8];
    Frag bf[4];
#pragma unroll
    for (int j = 0; j < 4; ++j) {
      const unsigned short* p = lb + (j * 16 + l16) * BSTRIDE;
      bf[j].u[0] = *(const v4u*)(p);
      bf[j].u[1] = *(const v4u*)(p + 16);
    }
#pragma unroll
    for (int j = 0; j < 4; ++j)
      acc[j] = __builtin_amdgcn_wmma_f32_16x16x32_bf16(
          false, af0.v, false, bf[j].v, (short)0, acc[j], false, false);
#pragma unroll
    for (int j = 0; j < 4; ++j)
      acc[4 + j] = __builtin_amdgcn_wmma_f32_16x16x32_bf16(
          false, af1.v, false, bf[j].v, (short)0, acc[4 + j], false, false);
  };

  const int ntiles = K >> 5;                      // >= 8 for all our shapes
  async_g2l_b128(lds0, gB);                       // tile 0 -> buf0
  async_g2l_b128(lds1, gB + 32);                  // tile 1 -> buf1

  // steady state: tiles 0 .. ntiles-3 (branch-free body)
#pragma unroll 2
  for (int t = 0; t < ntiles - 2; ++t) {
    asm volatile("s_wait_asynccnt 0x1" ::: "memory");  // tile t landed (in-order)
    __syncthreads();                                   // visible to all waves
    compute_tile(t);
    __syncthreads();                                   // all waves done with buf[t&1]
    async_g2l_b128((t & 1) ? lds1 : lds0, gB + (size_t)(t + 2) * 32);
  }
  // epilogue: tile ntiles-2 (one async still in flight), then tile ntiles-1
  asm volatile("s_wait_asynccnt 0x1" ::: "memory");
  __syncthreads();
  compute_tile(ntiles - 2);
  asm volatile("s_wait_asynccnt 0x0" ::: "memory");
  __syncthreads();
  compute_tile(ntiles - 1);

#pragma unroll
  for (int s = 0; s < 2; ++s) {
    const int orow = mbase + s * 16 + hi * 8;
#pragma unroll
    for (int j = 0; j < 4; ++j) {
      int col = nbase + j * 16 + l16;
#pragma unroll
      for (int r = 0; r < 8; ++r) {
        int row = orow + r;
        if (row < M) C[(size_t)row * NC + col] = acc[s * 4 + j][r];
      }
    }
  }
}

// ------------------------------------------------------------------
// edge MLP: a2[e,h] = gelu(ea @ We1 + be1) @ We2 + be2   (8 edges per block)
// wave32 shuffle reduction; only 2 barriers per edge
// ------------------------------------------------------------------
__global__ __launch_bounds__(256) void k_edge_mlp(
    const float* __restrict__ EA, const float* __restrict__ We1,
    const float* __restrict__ be1, const float* __restrict__ We2,
    const float* __restrict__ be2, float* __restrict__ a2, int E)
{
  __shared__ float ea_s[8 * 16];
  __shared__ v4f wred[8];
  const int tid = threadIdx.x;
  const int lane = tid & 31, wv = tid >> 5;
  const int ebase = blockIdx.x * 8;
  float w1[16];
#pragma unroll
  for (int f = 0; f < 16; ++f) w1[f] = We1[f * DIM + tid];
  float b1 = be1[tid];
  v4f w2 = *(const v4f*)(We2 + tid * 4);
  if (tid < 128) {
    int e = tid >> 4, f = tid & 15;
    int ei = ebase + e;
    ea_s[tid] = (ei < E) ? EA[(size_t)ei * 16 + f] : 0.f;
  }
  __syncthreads();
  for (int e = 0; e < 8; ++e) {
    float x = b1;
#pragma unroll
    for (int f = 0; f < 16; ++f) x = fmaf(ea_s[e * 16 + f], w1[f], x);
    float hv = gelu_tanh(x);
    v4f part = w2 * hv;
#pragma unroll
    for (int off = 16; off; off >>= 1) {   // wave32 reduce
      part.x += __shfl_xor(part.x, off, 32);
      part.y += __shfl_xor(part.y, off, 32);
      part.z += __shfl_xor(part.z, off, 32);
      part.w += __shfl_xor(part.w, off, 32);
    }
    if (lane == 0) wred[wv] = part;
    __syncthreads();
    if (tid == 0 && ebase + e < E) {
      v4f r = wred[0];
      for (int i = 1; i < 8; ++i) r += wred[i];
      float* o = a2 + (size_t)(ebase + e) * HEADS;
      o[0] = r.x + be2[0]; o[1] = r.y + be2[1];
      o[2] = r.z + be2[2]; o[3] = r.w + be2[3];
    }
    __syncthreads();
  }
}

// ------------------------------------------------------------------
// edge scores + segment max
// ------------------------------------------------------------------
__global__ void k_edge_score(const float* __restrict__ Out1, const float* __restrict__ a2,
                             const int* __restrict__ c2, const int* __restrict__ u2,
                             const float* __restrict__ bet2, float* __restrict__ s2,
                             float* __restrict__ mseg, int E) {
  int gid = blockIdx.x * blockDim.x + threadIdx.x;
  if (gid >= E * HEADS) return;
  int e = gid >> 2, h = gid & 3;
  int u = u2[e], c = c2[e];
  const float* q = Out1 + (size_t)u * NC1 + h * ZDIM;         // Q2[u]
  const float* k = Out1 + (size_t)c * NC1 + 256 + h * ZDIM;   // K2[c]
  float dot = 0.f;
  for (int z = 0; z < ZDIM; z += 4) {
    v4f a = *(const v4f*)(q + z); v4f b = *(const v4f*)(k + z);
    dot += a.x * b.x + a.y * b.y + a.z * b.z + a.w * b.w;
  }
  float b2 = fminf(softplus_f(bet2[0]), BETA_MAXV);
  float s = b2 * dot + a2[gid];
  s2[gid] = s;
  atomicMaxF(&mseg[u * HEADS + h], s);
}

__global__ void k_edge_exp(const int* __restrict__ u2, float* __restrict__ s2,
                           const float* __restrict__ mseg, float* __restrict__ den, int E) {
  int gid = blockIdx.x * blockDim.x + threadIdx.x;
  if (gid >= E * HEADS) return;
  int e = gid >> 2, h = gid & 3;
  float p = __expf(s2[gid] - mseg[u2[e] * HEADS + h]);
  s2[gid] = p;
  atomicAdd(&den[u2[e] * HEADS + h], p);
}

__global__ void k_edge_scatter(const float* __restrict__ Out1, const float* __restrict__ p2,
                               const float* __restrict__ den, const int* __restrict__ c2,
                               const int* __restrict__ u2, float* __restrict__ msgd,
                               float* __restrict__ msgs, int E) {
  int gid = blockIdx.x * blockDim.x + threadIdx.x;
  if (gid >= E * 64) return;
  int e = gid >> 6, q = gid & 63;
  int h = q >> 4, hz = q * 4;
  int u = u2[e], c = c2[e];
  float w = p2[(size_t)e * HEADS + h] / (den[u * HEADS + h] + 1e-9f);
  const float* kv = Out1 + (size_t)c * NC1 + 256 + hz;  // k2e
  const float* qv = Out1 + (size_t)u * NC1 + hz;        // q2e
#pragma unroll
  for (int i = 0; i < 4; ++i) {
    atomicAdd(&msgd[(size_t)u * DIM + hz + i], w * kv[i]);
    atomicAdd(&msgs[(size_t)c * DIM + hz + i], w * qv[i]);
  }
}

// ------------------------------------------------------------------
// pack GEMM2 A-matrix: [l2*msg_dst | l2*msg_src | lm*softmax_k(bm*sm)]
// ------------------------------------------------------------------
__global__ __launch_bounds__(256) void k_pack(const float* __restrict__ Out1,
                                              const float* __restrict__ msgd,
                                              const float* __restrict__ msgs,
                                              const float* __restrict__ lam2,
                                              const float* __restrict__ lamm,
                                              const float* __restrict__ betm,
                                              unsigned short* __restrict__ Apack) {
  int n = blockIdx.x, tid = threadIdx.x;
  float l2 = softplus_f(lam2[0]);
  unsigned short* row = Apack + (size_t)n * KPACK;
  row[tid]       = f2bf(l2 * msgd[(size_t)n * DIM + tid]);
  row[256 + tid] = f2bf(l2 * msgs[(size_t)n * DIM + tid]);
  if (tid < 128) {   // waves 0..3: one head (32 memory slots) per wave32
    float bm = fminf(softplus_f(betm[0]), BETA_MAXV);
    float lm = softplus_f(lamm[0]);
    float v = bm * Out1[(size_t)n * NC1 + 512 + tid];
    float mx = v;
    for (int off = 16; off; off >>= 1) mx = fmaxf(mx, __shfl_xor(mx, off, 32));
    float ev = __expf(v - mx);
    float sum = ev;
    for (int off = 16; off; off >>= 1) sum += __shfl_xor(sum, off, 32);
    row[512 + tid] = f2bf(lm * (ev / sum));
  }
}

// ------------------------------------------------------------------
// finalize: add motif term, clip grad, damped update, clip state
// ------------------------------------------------------------------
__global__ __launch_bounds__(256) void k_final(const float* __restrict__ G,
                                               const float* __restrict__ gradl,
                                               const float* __restrict__ Out1,
                                               const float* __restrict__ Uq,
                                               const float* __restrict__ Uk,
                                               const float* __restrict__ lam3,
                                               const float* __restrict__ bet3,
                                               float* __restrict__ out) {
  __shared__ float coef[16];
  __shared__ float red[256];
  int n = blockIdx.x, d = threadIdx.x;
  if (d < 8) {                       // (qk, h) pairs; softmax over T=2 motif types
    int qk = d >> 2, h = d & 3;
    float l3 = softplus_f(lam3[0]);
    float b3 = fminf(softplus_f(bet3[0]), BETA_MAXV);
    const float* sc = Out1 + (size_t)n * NC1 + 640 + qk * 8 + h * 2;
    float s0 = b3 * sc[0], s1 = b3 * sc[1];
    float mx = fmaxf(s0, s1);
    float e0 = __expf(s0 - mx), e1 = __expf(s1 - mx);
    float inv = l3 / (e0 + e1);
    coef[d * 2]     = e0 * inv;
    coef[d * 2 + 1] = e1 * inv;
  }
  __syncthreads();
  float macc = 0.f;
#pragma unroll
  for (int h = 0; h < 4; ++h)
#pragma unroll
    for (int t = 0; t < 2; ++t) {
      macc += coef[h * 2 + t]     * Uq[(t * 4 + h) * DIM + d];
      macc += coef[8 + h * 2 + t] * Uk[(t * 4 + h) * DIM + d];
    }
  float grad = -(gradl[(size_t)n * DIM + d] + macc);
  red[d] = grad * grad; __syncthreads();
  for (int s = 128; s; s >>= 1) { if (d < s) red[d] += red[d + s]; __syncthreads(); }
  float sc1 = 1.0f / fmaxf(sqrtf(red[0]), 1.0f);            // grad clip, c=1
  float gn = G[(size_t)n * DIM + d] - DAMP * grad * sc1;
  __syncthreads();
  red[d] = gn * gn; __syncthreads();
  for (int s = 128; s; s >>= 1) { if (d < s) red[d] += red[d + s]; __syncthreads(); }
  float sc2 = 10.0f / fmaxf(sqrtf(red[0]), 10.0f);          // state clip, c=10
  out[(size_t)n * DIM + d] = gn * sc2;
}

// ------------------------------------------------------------------
extern "C" void kernel_launch(void* const* d_in, const int* in_sizes, int n_in,
                              void* d_out, int out_size, void* d_ws, size_t ws_size,
                              hipStream_t stream) {
  const float* G    = (const float*)d_in[0];
  const float* EA   = (const float*)d_in[1];
  const int*   c2   = (const int*)d_in[2];
  const int*   u2   = (const int*)d_in[3];
  const float* Wq2  = (const float*)d_in[4];
  const float* Wk2  = (const float*)d_in[5];
  const float* Wq3  = (const float*)d_in[6];
  const float* Wk3  = (const float*)d_in[7];
  const float* Ttau = (const float*)d_in[8];
  const float* Wqm  = (const float*)d_in[9];
  const float* Wkm  = (const float*)d_in[10];
  const float* Bmem = (const float*)d_in[11];
  const float* We1  = (const float*)d_in[12];
  const float* be1  = (const float*)d_in[13];
  const float* We2  = (const float*)d_in[14];
  const float* be2  = (const float*)d_in[15];
  const float* lng  = (const float*)d_in[16];
  const float* lnb  = (const float*)d_in[17];
  const float* lam2 = (const float*)d_in[18];
  const float* lam3 = (const float*)d_in[19];
  const float* lamm = (const float*)d_in[20];
  const float* bet2 = (const float*)d_in[21];
  const float* bet3 = (const float*)d_in[22];
  const float* betm = (const float*)d_in[23];

  char* w = (char*)d_ws;
  size_t off = 0;
  auto alloc = [&](size_t bytes) -> char* {
    char* p = w + off;
    off += (bytes + 255) & ~(size_t)255;
    return p;
  };
  unsigned short* gbf    = (unsigned short*)alloc((size_t)N_NODES * DIM * 2);
  unsigned short* Wcat1  = (unsigned short*)alloc((size_t)NC1 * DIM * 2);
  unsigned short* Wcat2T = (unsigned short*)alloc((size_t)DIM * KPACK * 2);
  float* Km    = (float*)alloc(8192 * 4);
  float* Uq    = (float*)alloc(2048 * 4);
  float* Uk    = (float*)alloc(2048 * 4);
  float* Out1  = (float*)alloc((size_t)N_NODES * NC1 * 4);
  float* a2    = (float*)alloc((size_t)EDGES * HEADS * 4);
  float* p2    = (float*)alloc((size_t)EDGES * HEADS * 4);
  float* mseg  = (float*)alloc((size_t)N_NODES * HEADS * 4);
  float* den   = (float*)alloc((size_t)N_NODES * HEADS * 4);
  float* msgd  = (float*)alloc((size_t)N_NODES * DIM * 4);
  float* msgs  = (float*)alloc((size_t)N_NODES * DIM * 4);
  unsigned short* Apack = (unsigned short*)alloc((size_t)N_NODES * KPACK * 2);
  float* gradl = (float*)alloc((size_t)N_NODES * DIM * 4);

  hipMemsetAsync(den,  0, (size_t)N_NODES * HEADS * 4, stream);
  hipMemsetAsync(msgd, 0, (size_t)N_NODES * DIM * 4, stream);
  hipMemsetAsync(msgs, 0, (size_t)N_NODES * DIM * 4, stream);
  hipMemsetAsync(Wcat1 + (size_t)656 * DIM, 0, (size_t)48 * DIM * 2, stream);  // pad cols

  k_fillf<<<(N_NODES * HEADS + 255) / 256, 256, 0, stream>>>(mseg, -INFINITY, N_NODES * HEADS);
  k_cvtW<<<256, 256, 0, stream>>>(Wq2, Wcat1, Wcat2T, 0);
  k_cvtW<<<256, 256, 0, stream>>>(Wk2, Wcat1, Wcat2T, 256);
  k_km<<<32, 256, 0, stream>>>(Bmem, Wkm, Km);
  k_u<<<16, 256, 0, stream>>>(Ttau, Wq3, Wk3, Uq, Uk, Wcat1);
  k_m2p<<<128, 256, 0, stream>>>(Km, Wqm, Wcat1, Wcat2T);
  k_layernorm<<<N_NODES, 256, 0, stream>>>(G, lng, lnb, gbf);
  k_gemm_bf16<<<dim3((N_NODES + 255) / 256, NC1 / 64), 256, 0, stream>>>(
      gbf, Wcat1, Out1, N_NODES, DIM, NC1);
  k_edge_mlp<<<(EDGES + 7) / 8, 256, 0, stream>>>(EA, We1, be1, We2, be2, a2, EDGES);
  k_edge_score<<<(EDGES * HEADS + 255) / 256, 256, 0, stream>>>(
      Out1, a2, c2, u2, bet2, p2, mseg, EDGES);
  k_edge_exp<<<(EDGES * HEADS + 255) / 256, 256, 0, stream>>>(u2, p2, mseg, den, EDGES);
  k_edge_scatter<<<(EDGES * 64 + 255) / 256, 256, 0, stream>>>(
      Out1, p2, den, c2, u2, msgd, msgs, EDGES);
  k_pack<<<N_NODES, 256, 0, stream>>>(Out1, msgd, msgs, lam2, lamm, betm, Apack);
  k_gemm_bf16<<<dim3((N_NODES + 255) / 256, DIM / 64), 256, 0, stream>>>(
      Apack, Wcat2T, gradl, N_NODES, KPACK, DIM);
  k_final<<<N_NODES, 256, 0, stream>>>(G, gradl, Out1, Uq, Uk, lam3, bet3, (float*)d_out);
}
